// ExodusNet_74268574482490
// MI455X (gfx1250) — matrix operand order repaired and checked by reading, not compiled, not run on hardware
//
#include <hip/hip_runtime.h>

// CDNA5 / gfx1250: wave32, WMMA f32 <- f16 x f16 (16x16x32), fp32-accurate via
// two-term f16 split of both operands (3 WMMAs: Wh*xh + Wh*xl + Wl*xh).

typedef __attribute__((ext_vector_type(16))) _Float16 v16h;
typedef __attribute__((ext_vector_type(8)))  float    v8f;

#define WAVES_PER_BLOCK 2
#define B_PER_WAVE      32
#define F_DIM           32
#define T_DIM           100
#define TPAD            104                 // lds_x row stride (dwords), 16B-aligned rows
#define XSKEW           16                  // +64B skew for rows f>=16 -> disjoint LDS banks per lane-half
#define WROWS           101                 // weighted/spike row stride (odd -> conflict-free serial scan)
#define LDSX            (F_DIM * TPAD + XSKEW)      // 3344 floats
#define LDSW            (B_PER_WAVE * WROWS)        // 3232 floats
#define WAVE_LDS        (LDSX + LDSW)               // 6576 floats

__global__ __launch_bounds__(WAVES_PER_BLOCK * 32)
void ExodusNet_lif_kernel(const float* __restrict__ x,
                          const float* __restrict__ W,
                          float* __restrict__ out) {
    __shared__ float smem[WAVES_PER_BLOCK * WAVE_LDS];   // 52,608 B

    const int tid  = threadIdx.x;
    const int wave = tid >> 5;
    const int lane = tid & 31;
    float* lx = smem + wave * WAVE_LDS;       // staged x[b] slab, f-major
    float* lw = lx + LDSX;                    // weighted -> spikes, [32][101]
    const int b0 = (blockIdx.x * WAVES_PER_BLOCK + wave) * B_PER_WAVE;

    const int h2 = (lane >> 4) & 1;           // lane half (selects K range)
    const int nn = lane & 15;                 // N index (timestep within tile)

    // ---- Build A fragments: A[m,k] = W[k] for all m (all D rows identical).
    // A 16x32 f16 layout: element e -> K = e + 16*h2.
    v16h ah, al;
#pragma unroll
    for (int e = 0; e < 16; ++e) {
        const int k = e + (h2 ? 16 : 0);
        const float wf = W[k];
        const _Float16 wh = (_Float16)wf;
        ah[e] = wh;
        al[e] = (_Float16)(wf - (float)wh);   // f16 residual of W
    }

    for (int i = 0; i < B_PER_WAVE; ++i) {
        const size_t b = (size_t)(b0 + i);

        // ---- Stage x[b] (32x100 f32) into LDS with fully coalesced b128 loads.
        // flat float4 chunk idx in [0,800): f = idx/25, r = idx%25 (row = 25 float4s).
        const float4* xb = (const float4*)(x + b * (size_t)(F_DIM * T_DIM));
#pragma unroll
        for (int c = 0; c < 25; ++c) {
            const int idx = c * 32 + lane;
            const float4 vv = xb[idx];
            const int f = idx / 25;
            const int r = idx - f * 25;
            float* dst = lx + f * TPAD + ((f >= 16) ? XSKEW : 0) + r * 4;
            *(float4*)dst = vv;               // 16B-aligned ds_store_b128
        }
        __syncthreads();

        // ---- Projection: weighted[b, t] = sum_f W[f]*x[b,f,t] via WMMA.
        // B 32x16 f16 layout: n = lane&15, element e -> K = e + 16*h2.
        const float* src = lx + h2 * (16 * TPAD + XSKEW);
#pragma unroll
        for (int t0 = 0; t0 < 112; t0 += 16) {        // 7 tiles cover T=100
            const int t  = t0 + nn;
            const int tt = (t < T_DIM) ? t : 0;        // clamp tail (garbage cols unused)
            const float* s = src + tt;
            v16h bh, bl;
#pragma unroll
            for (int e = 0; e < 16; ++e) {
                const float xv = s[e * TPAD];          // conflict-free ds_load_b32
                const _Float16 xh = (_Float16)xv;
                bh[e] = xh;
                bl[e] = (_Float16)(xv - (float)xh);    // f16 residual of x
            }
            v8f acc = {};
            acc = __builtin_amdgcn_wmma_f32_16x16x32_f16(false, ah, false, bh,
                                                         (short)0, acc, false, false);
            acc = __builtin_amdgcn_wmma_f32_16x16x32_f16(false, ah, false, bl,
                                                         (short)0, acc, false, false);
            acc = __builtin_amdgcn_wmma_f32_16x16x32_f16(false, al, false, bh,
                                                         (short)0, acc, false, false);
            if ((h2 == 0) && (t < T_DIM)) lw[i * WROWS + t] = acc[0];
        }
        __syncthreads();
    }

    // ---- Serial LIF scan: one neuron per lane (the only true serial chain).
    {
        const float kA = 0.90483741803595952f;   // exp(-1/tau), tau = 10
        const float kB = 0.09516258196404048f;   // 1 - alpha (norm_input)
        float v = 0.0f;
        float* row = lw + lane * WROWS;
#pragma unroll 4
        for (int t = 0; t < T_DIM; ++t) {
            const float wv = row[t];
            v = kA * v + kB * wv;
            const float sp = (v >= 1.0f) ? 1.0f : 0.0f;  // SingleSpike, thr = 1
            v -= sp;                                      // MembraneSubtract
            row[t] = sp;
        }
    }
    __syncthreads();

    // ---- Coalesced spike writeout: 3200 contiguous floats per wave.
    float* ob = out + (size_t)b0 * T_DIM;
#pragma unroll 4
    for (int q = 0; q < 100; ++q) {
        const int idx = q * 32 + lane;
        const int j = idx / T_DIM;
        const int t = idx - j * T_DIM;
        ob[idx] = lw[j * WROWS + t];
    }
}

extern "C" void kernel_launch(void* const* d_in, const int* in_sizes, int n_in,
                              void* d_out, int out_size, void* d_ws, size_t ws_size,
                              hipStream_t stream) {
    (void)n_in; (void)out_size; (void)d_ws; (void)ws_size;
    const float* x = (const float*)d_in[0];   // [B, 2, 4, 4, 100] f32 = [B, 32, 100]
    const float* W = (const float*)d_in[1];   // [1, 32] f32
    float* out = (float*)d_out;               // [B, 1, 100] f32 spikes

    const int B = in_sizes[0] / (F_DIM * T_DIM);          // 32768
    const int grid = B / (WAVES_PER_BLOCK * B_PER_WAVE);  // 512 blocks of 64 threads
    ExodusNet_lif_kernel<<<grid, WAVES_PER_BLOCK * 32, 0, stream>>>(x, W, out);
}
